// DynamicCellEvolver_29334626632200
// MI455X (gfx1250) — matrix operand order repaired and checked by reading, not compiled, not run on hardware
//
#include <hip/hip_runtime.h>
#include <hip/hip_bf16.h>

// Problem constants (from reference)
#define B_    128
#define T_    24
#define D_    8088
#define L_    128
#define H_    256
#define OBS_  3
#define PRED_ 22   // T - OBS + 1

typedef __attribute__((ext_vector_type(16))) __bf16 v16bf;
typedef __attribute__((ext_vector_type(8)))  float  v8f;

// ---------------------------------------------------------------------------
// WMMA wrapper: D = A(16x32 bf16) x B(32x16 bf16) + C(16x16 f32)
// ---------------------------------------------------------------------------
__device__ __forceinline__ v8f wmma_bf16(v16bf a, v16bf b, v8f c) {
  return __builtin_amdgcn_wmma_f32_16x16x32_bf16(
      /*neg_a=*/false, a, /*neg_b=*/false, b,
      /*c_mod=*/(short)0, c, /*reuse_a=*/false, /*reuse_b=*/false);
}

// ---------------------------------------------------------------------------
// Fragment loaders (layouts per CDNA5 ISA 7.12.2, wave32)
// A 16x32 bf16: lane m=lane&15, half=lane>>4; elem i -> K = i + (i>=8?8:0) + 8*half
// B 32x16 bf16 (from row-major W[N][K], i.e. B = W^T): lane n=lane&15;
//              elem i -> K = i + (i>=8?8:0) + 8*half
// C/D 16x16 f32: elem r -> row = r + 8*(lane>=16), col = lane&15
// ---------------------------------------------------------------------------
__device__ __forceinline__ v16bf load_a_f32(const float* __restrict__ base, int lda) {
  const int lane = threadIdx.x & 31;
  const int m = lane & 15, h = lane >> 4;
  const float* p = base + (long)m * lda + h * 8;
  v16bf a;
#pragma unroll
  for (int i = 0; i < 8; ++i) a[i]     = (__bf16)p[i];
#pragma unroll
  for (int i = 0; i < 8; ++i) a[8 + i] = (__bf16)p[16 + i];
  return a;
}

// Branch-free K-tail guard: clamp address (load unconditional), select value.
__device__ __forceinline__ v16bf load_a_f32_guard(const float* __restrict__ base,
                                                  int lda, int kvalid) {
  const int lane = threadIdx.x & 31;
  const int m = lane & 15, h = lane >> 4;
  const float* p = base + (long)m * lda;
  v16bf a;
#pragma unroll
  for (int i = 0; i < 16; ++i) {
    const int k   = (i < 8 ? i : i + 8) + h * 8;
    const int kcl = (k < kvalid) ? k : 0;
    const float v = p[kcl];
    a[i] = (k < kvalid) ? (__bf16)v : (__bf16)0.0f;
  }
  return a;
}

__device__ __forceinline__ v16bf load_a_bf16(const __bf16* __restrict__ base, int lda) {
  const int lane = threadIdx.x & 31;
  const int m = lane & 15, h = lane >> 4;
  const __bf16* p = base + (long)m * lda + h * 8;
  v16bf a;
#pragma unroll
  for (int i = 0; i < 8; ++i) a[i]     = p[i];
#pragma unroll
  for (int i = 0; i < 8; ++i) a[8 + i] = p[16 + i];
  return a;
}

// LDS A fragment, fixed lda = 136 bf16 (padded: conflict-free ds_load_b128)
__device__ __forceinline__ v16bf load_a_lds_fwd(const __bf16* base) {
  const int lane = threadIdx.x & 31;
  const int m = lane & 15, h = lane >> 4;
  const __bf16* p = base + m * 136 + h * 8;
  v16bf a;
#pragma unroll
  for (int i = 0; i < 8; ++i) a[i]     = p[i];
#pragma unroll
  for (int i = 0; i < 8; ++i) a[8 + i] = p[16 + i];
  return a;
}

__device__ __forceinline__ v16bf load_bt_bf16(const __bf16* __restrict__ base, int ldk) {
  const int lane = threadIdx.x & 31;
  const int n = lane & 15, h = lane >> 4;
  const __bf16* p = base + (long)n * ldk + h * 8;
  v16bf b;
#pragma unroll
  for (int i = 0; i < 8; ++i) b[i]     = p[i];
#pragma unroll
  for (int i = 0; i < 8; ++i) b[8 + i] = p[16 + i];
  return b;
}

__device__ __forceinline__ v16bf load_bt_bf16_guard(const __bf16* __restrict__ base,
                                                    int ldk, int kvalid, int nvalid) {
  const int lane = threadIdx.x & 31;
  const int n = lane & 15, h = lane >> 4;
  const bool nok = n < nvalid;
  const __bf16* p = base + (long)(nok ? n : 0) * ldk;
  v16bf b;
#pragma unroll
  for (int i = 0; i < 16; ++i) {
    const int k   = (i < 8 ? i : i + 8) + h * 8;
    const int kcl = (k < kvalid) ? k : 0;
    const __bf16 v = p[kcl];
    b[i] = (nok && k < kvalid) ? v : (__bf16)0.0f;
  }
  return b;
}

__device__ __forceinline__ float fsigmoid(float x) {
  return 1.0f / (1.0f + __expf(-x));
}

// ===========================================================================
// Kernel 0: one-pass f32 -> bf16 conversion of all weight matrices into ws.
// Order in dst: proj_W, rec_W, Wih, Whh, Wz, ode_W1, ode_W2, gp_W1, gp_W2.
// ===========================================================================
#define CVT_TOT 2480128L
__global__ __launch_bounds__(256) void k_cvt(
    const float* __restrict__ s0, const float* __restrict__ s1,
    const float* __restrict__ s2, const float* __restrict__ s3,
    const float* __restrict__ s4, const float* __restrict__ s5,
    const float* __restrict__ s6, const float* __restrict__ s7,
    const float* __restrict__ s8, __bf16* __restrict__ dst)
{
  const long stride = (long)gridDim.x * 256;
  for (long i = (long)blockIdx.x * 256 + threadIdx.x; i < CVT_TOT; i += stride) {
    long j = i;
    const float* s;
    if (j < 1035264)                   s = s0;   // proj_W 128x8088
    else if ((j -= 1035264) < 1035264) s = s1;   // rec_W 8088x128
    else if ((j -= 1035264) < 98304)   s = s2;   // Wih 768x128
    else if ((j -= 98304)  < 196608)   s = s3;   // Whh 768x256
    else if ((j -= 196608) < 32768)    s = s4;   // Wz 128x256
    else if ((j -= 32768)  < 32768)    s = s5;   // ode_W1 256x128
    else if ((j -= 32768)  < 32768)    s = s6;   // ode_W2 128x256
    else if ((j -= 32768)  < 8192)     s = s7;   // gp_W1 64x128
    else { j -= 8192;                  s = s8; } // gp_W2 128x64
    dst[i] = (__bf16)s[j];
  }
}

// ===========================================================================
// Kernel 1: x_obs partials = z_seq[:, :OBS, :] @ proj_W^T  (384x128, K=8088)
// Split-K x4 for parallelism; bf16 B; f32 partials reduced by k_proj_reduce.
// Grid: (24 row tiles, 4 K-splits) x 256 thr (8 waves = 8 col tiles).
// ===========================================================================
__global__ __launch_bounds__(256) void k_proj(const float* __restrict__ z_seq,
                                              const __bf16* __restrict__ Wpb,
                                              float* __restrict__ part) {
  const int wave = threadIdx.x >> 5;
  const int lane = threadIdx.x & 31;
  const int nl = lane & 15, mhalf = (lane >> 4) * 8;
  const int rt = blockIdx.x;          // 0..23 over M=384 (t-major, b-minor)
  const int ks = blockIdx.y;          // 0..3  K split
  const int ct = wave;                // 0..7  over N=128
  const int t  = rt >> 3;
  const int b0 = (rt & 7) << 4;
  const float*  Abase = z_seq + ((long)b0 * T_ + t) * D_;   // row stride T_*D_
  const __bf16* Bbase = Wpb + (long)(ct * 16) * D_;

  // 253 chunks total (252 full + 1 tail of 24); split [ks*64, min(253, +64))
  const int c0 = ks * 64;
  const int c1 = (c0 + 64 < 253) ? (c0 + 64) : 253;
  const int cfull = (c1 < 252) ? c1 : 252;

  v8f c = {};
  for (int ch = c0; ch < cfull; ++ch) {
    const int kc = ch * 32;
    __builtin_prefetch(Abase + kc + 256, 0, 1);   // stream A ahead
    v16bf a = load_a_f32(Abase + kc, T_ * D_);
    v16bf b = load_bt_bf16(Bbase + kc, D_);
    c = wmma_bf16(a, b, c);
  }
  if (c1 == 253) {                                 // K tail: 8088 - 8064 = 24
    const int kc = 252 * 32;
    v16bf a = load_a_f32_guard(Abase + kc, T_ * D_, D_ - kc);
    v16bf b = load_bt_bf16_guard(Bbase + kc, D_, D_ - kc, 16);
    c = wmma_bf16(a, b, c);
  }
  const int col = ct * 16 + nl;
#pragma unroll
  for (int e = 0; e < 8; ++e) {
    const int row = rt * 16 + mhalf + e;           // == t*128 + b
    part[(long)ks * (384 * L_) + row * L_ + col] = c[e];
  }
}

__global__ __launch_bounds__(256) void k_proj_reduce(const float* __restrict__ part,
                                                     const float* __restrict__ bp,
                                                     float* __restrict__ xobs) {
  const int idx = blockIdx.x * 256 + threadIdx.x;  // 192 blocks -> 49152
  float s = bp[idx & (L_ - 1)];
#pragma unroll
  for (int ks = 0; ks < 4; ++ks) s += part[(long)ks * (384 * L_) + idx];
  xobs[idx] = s;
}

// ===========================================================================
// Kernel 2: persistent single-workgroup GRU(3) + z0 + RK4(21). 32 waves on
// one WGP. All weight B-fragments are pre-converted bf16; RK4 state feeding
// the A side (zcur/zin) is mirrored in bf16 so the serial critical path has
// no f32->bf16 converts. Stage-input prep is fused into the GEMM2 epilogue
// (each wave owns exactly the elements it writes) -> 4 barriers per stage.
// ===========================================================================
__global__ __launch_bounds__(1024) void k_recurrent(
    const float* __restrict__ xobs,
    const __bf16* __restrict__ Wihb, const __bf16* __restrict__ Whhb,
    const float* __restrict__ bih,   const float* __restrict__ bhh,
    const __bf16* __restrict__ Wzb,  const float* __restrict__ bz,
    const __bf16* __restrict__ W1b,  const float* __restrict__ b1,
    const __bf16* __restrict__ W2b,  const float* __restrict__ b2,
    float* __restrict__ hA, float* __restrict__ hB,
    float* __restrict__ zcur_f, __bf16* __restrict__ zcur_b,
    __bf16* __restrict__ zin_b,
    float* __restrict__ k1, float* __restrict__ k2, float* __restrict__ k3,
    float* __restrict__ zmean)
{
  __shared__ __bf16 u_lds[128 * 136];   // 34816 B: one K-half of tanh(z@W1^T+b1)
  const int wave = threadIdx.x >> 5;
  const int lane = threadIdx.x & 31;
  const int nl = lane & 15, mhalf = (lane >> 4) * 8;

  float* hcur = hA;
  float* hnxt = hB;

  // ------------------------- GRU over OBS_ steps ---------------------------
  for (int t = 0; t < OBS_; ++t) {
    const float* x = xobs + t * (B_ * L_);
#pragma unroll 1
    for (int task = wave * 4; task < wave * 4 + 4; ++task) {  // 128 tasks
      const int rt  = task >> 4;   // 0..7  rows of B_
      const int ctj = task & 15;   // 0..15 cols of H_
      v8f ci[3] = {{}, {}, {}};
      v8f ch[3] = {{}, {}, {}};
#pragma unroll
      for (int kc = 0; kc < L_; kc += 32) {        // gi = x @ Wih^T
        v16bf a = load_a_f32(x + rt * 16 * L_ + kc, L_);
#pragma unroll
        for (int g = 0; g < 3; ++g) {
          v16bf b = load_bt_bf16(Wihb + (long)(g * H_ + ctj * 16) * L_ + kc, L_);
          ci[g] = wmma_bf16(a, b, ci[g]);
        }
      }
      if (t > 0) {                                 // gh = h @ Whh^T
#pragma unroll
        for (int kc = 0; kc < H_; kc += 32) {
          v16bf a = load_a_f32(hcur + rt * 16 * H_ + kc, H_);
#pragma unroll
          for (int g = 0; g < 3; ++g) {
            v16bf b = load_bt_bf16(Whhb + (long)(g * H_ + ctj * 16) * H_ + kc, H_);
            ch[g] = wmma_bf16(a, b, ch[g]);
          }
        }
      }
      const int j = ctj * 16 + nl;
      const float bir = bih[0 * H_ + j], biz = bih[1 * H_ + j], bin = bih[2 * H_ + j];
      const float bhr = bhh[0 * H_ + j], bhz = bhh[1 * H_ + j], bhn = bhh[2 * H_ + j];
#pragma unroll
      for (int e = 0; e < 8; ++e) {
        const int row = rt * 16 + mhalf + e;
        const float r  = fsigmoid((ci[0][e] + bir) + (ch[0][e] + bhr));
        const float zg = fsigmoid((ci[1][e] + biz) + (ch[1][e] + bhz));
        const float n  = tanhf((ci[2][e] + bin) + r * (ch[2][e] + bhn));
        const float ho = (t > 0) ? hcur[row * H_ + j] : 0.0f;
        hnxt[row * H_ + j] = (1.0f - zg) * n + zg * ho;
      }
    }
    __syncthreads();
    float* tmp = hcur; hcur = hnxt; hnxt = tmp;
  }

  // ------------------------- z0 = h @ Wz^T + bz ----------------------------
#pragma unroll 1
  for (int o = wave * 2; o < wave * 2 + 2; ++o) {   // 64 tiles of 128x128
    const int rt = o >> 3, ct = o & 7;
    v8f c = {};
#pragma unroll
    for (int kc = 0; kc < H_; kc += 32) {
      v16bf a = load_a_f32(hcur + rt * 16 * H_ + kc, H_);
      v16bf b = load_bt_bf16(Wzb + (long)(ct * 16) * H_ + kc, H_);
      c = wmma_bf16(a, b, c);
    }
    const int col = ct * 16 + nl;
    const float bv = bz[col];
#pragma unroll
    for (int e = 0; e < 8; ++e) {
      const int row = rt * 16 + mhalf + e;
      const float v = c[e] + bv;
      zcur_f[row * L_ + col] = v;
      zcur_b[row * L_ + col] = (__bf16)v;
      zmean[(long)row * (PRED_ * L_) + col] = v;    // p = 0
    }
  }
  __syncthreads();

  // ------------------------- RK4: 21 steps ---------------------------------
  const float DT = 1.0f / (float)(PRED_ - 1);       // linspace(0,1,22) diffs
  for (int p = 1; p < PRED_; ++p) {
#pragma unroll 1
    for (int s = 0; s < 4; ++s) {
      const __bf16* zsrc = (s == 0) ? zcur_b : zin_b;
      // f(zsrc) = tanh(zsrc @ W1^T + b1) @ W2^T + b2; GEMM2 K split in two
      v8f acc[2] = {{}, {}};
#pragma unroll 1
      for (int ph = 0; ph < 2; ++ph) {
        // GEMM1 half: u cols [ph*128, ph*128+128) -> LDS (bf16)
#pragma unroll 1
        for (int i = 0; i < 2; ++i) {
          const int tsk = wave * 2 + i;             // 0..63
          const int rt = tsk >> 3, ctl = tsk & 7;
          const int col0 = ph * 128 + ctl * 16;     // W1 row index
          v8f c = {};
#pragma unroll
          for (int kc = 0; kc < L_; kc += 32) {
            v16bf a = load_a_bf16(zsrc + rt * 16 * L_ + kc, L_);
            v16bf b = load_bt_bf16(W1b + (long)col0 * L_ + kc, L_);
            c = wmma_bf16(a, b, c);
          }
          const float bv = b1[col0 + nl];
#pragma unroll
          for (int e = 0; e < 8; ++e) {
            const int row = rt * 16 + mhalf + e;
            u_lds[row * 136 + ctl * 16 + nl] = (__bf16)tanhf(c[e] + bv);
          }
        }
        __syncthreads();
        // GEMM2 partial accumulate over this K half
#pragma unroll 1
        for (int i = 0; i < 2; ++i) {
          const int o = wave * 2 + i;
          const int rt = o >> 3, ct = o & 7;
#pragma unroll
          for (int kc = 0; kc < 128; kc += 32) {
            v16bf a = load_a_lds_fwd(u_lds + rt * 16 * 136 + kc);
            v16bf b = load_bt_bf16(W2b + (long)(ct * 16) * H_ + ph * 128 + kc, H_);
            acc[i] = wmma_bf16(a, b, acc[i]);
          }
        }
        if (ph == 0) __syncthreads();               // protect LDS reuse
      }
      // Epilogue: write k_s and FUSE next-stage input prep (own elements only)
#pragma unroll 1
      for (int i = 0; i < 2; ++i) {
        const int o = wave * 2 + i;
        const int rt = o >> 3, ct = o & 7;
        const int col = ct * 16 + nl;
        const float bv = b2[col];
#pragma unroll
        for (int e = 0; e < 8; ++e) {
          const int row = rt * 16 + mhalf + e;
          const int idx = row * L_ + col;
          const float kv = acc[i][e] + bv;
          if (s == 0) {
            k1[idx] = kv;
            zin_b[idx] = (__bf16)(zcur_f[idx] + (DT / 3.0f) * kv);
          } else if (s == 1) {
            k2[idx] = kv;
            zin_b[idx] = (__bf16)(zcur_f[idx] + DT * (kv - k1[idx] * (1.0f / 3.0f)));
          } else if (s == 2) {
            k3[idx] = kv;
            zin_b[idx] = (__bf16)(zcur_f[idx] + DT * (k1[idx] - k2[idx] + kv));
          } else {
            const float zn = zcur_f[idx] +
                DT * (k1[idx] + 3.0f * (k2[idx] + k3[idx]) + kv) * 0.125f;
            zcur_f[idx] = zn;
            zcur_b[idx] = (__bf16)zn;
            zmean[(long)row * (PRED_ * L_) + p * L_ + col] = zn;
          }
        }
      }
      __syncthreads();
    }
  }
}

// ===========================================================================
// Kernel 3: logvar = relu(zm@gpW1^T+b1)@gpW2^T+b2 ; z_sample (bf16) =
// zm + eps*exp(.5*lv). One 16-row tile per wave; hidden staged in LDS.
// ===========================================================================
__global__ __launch_bounds__(256) void k_gp_sample(
    const float* __restrict__ zmean, const float* __restrict__ eps,
    const __bf16* __restrict__ W1b, const float* __restrict__ b1,
    const __bf16* __restrict__ W2b, const float* __restrict__ b2,
    __bf16* __restrict__ zsamp_b, float* __restrict__ logvar_out)
{
  __shared__ __bf16 hid[8 * 16 * 72];   // per-wave 16x64 (+pad)
  const int wave = threadIdx.x >> 5;
  const int lane = threadIdx.x & 31;
  const int nl = lane & 15, mhalf = (lane >> 4) * 8;
  const int rt = blockIdx.x * 8 + wave;       // 0..175
  const float* Ab = zmean + (long)rt * 16 * L_;
  __bf16* hw = hid + wave * (16 * 72);

  // hidden = relu(zm @ W1^T + b1) : 16 x 64
#pragma unroll 1
  for (int ct = 0; ct < 4; ++ct) {
    v8f c = {};
#pragma unroll
    for (int kc = 0; kc < L_; kc += 32) {
      v16bf a = load_a_f32(Ab + kc, L_);
      v16bf b = load_bt_bf16(W1b + (long)(ct * 16) * L_ + kc, L_);
      c = wmma_bf16(a, b, c);
    }
    const float bv = b1[ct * 16 + nl];
#pragma unroll
    for (int e = 0; e < 8; ++e) {
      const float v = c[e] + bv;
      hw[(mhalf + e) * 72 + ct * 16 + nl] = (__bf16)(v > 0.0f ? v : 0.0f);
    }
  }
  __syncthreads();
  // logvar = hidden @ W2^T + b2 : 16 x 128; then sample
#pragma unroll 1
  for (int ct = 0; ct < 8; ++ct) {
    v8f c = {};
#pragma unroll
    for (int kc = 0; kc < 64; kc += 32) {
      v16bf a = load_a_bf16(hw + kc, 72);
      v16bf b = load_bt_bf16(W2b + (long)(ct * 16) * 64 + kc, 64);
      c = wmma_bf16(a, b, c);
    }
    const int col = ct * 16 + nl;
    const float bv = b2[col];
#pragma unroll
    for (int e = 0; e < 8; ++e) {
      const long idx = (long)(rt * 16 + mhalf + e) * L_ + col;
      const float lv = c[e] + bv;
      logvar_out[idx] = lv;
      zsamp_b[idx] = (__bf16)(zmean[idx] + eps[idx] * __expf(0.5f * lv));
    }
  }
}

// ===========================================================================
// Kernel 4: z_pred = z_sample @ rec_W^T + rec_b  (2816 x 8088, K=128).
// Store-bandwidth bound (~91 MB -> ~4 us at 23.3 TB/s). All-bf16 fragment
// loads (zsamp_b 0.7 MB + rec_Wb 2 MB are L2-resident), zero converts.
// N tail (8088 = 505*16 + 8): branch-free clamped loads, predicated stores.
// ===========================================================================
__global__ __launch_bounds__(256) void k_rec(
    const __bf16* __restrict__ zsamp_b, const __bf16* __restrict__ Wb,
    const float* __restrict__ bias, float* __restrict__ zpred)
{
  const int wave = threadIdx.x >> 5;
  const int lane = threadIdx.x & 31;
  const int nl = lane & 15, mhalf = (lane >> 4) * 8;
  const int ct = blockIdx.x * 8 + wave;      // col tile over N=8088
  if (ct * 16 >= D_) return;                 // wave-uniform exit
  const int rb = blockIdx.y;                 // 0..43 -> rows rb*64
  const int n0 = ct * 16;
  const int nvalid = (D_ - n0 < 16) ? (D_ - n0) : 16;

  v8f acc[4] = {{}, {}, {}, {}};
  const __bf16* Ab = zsamp_b + (long)rb * 64 * L_;
  const __bf16* Bb = Wb + (long)n0 * L_;

  if (nvalid == 16) {
#pragma unroll
    for (int kc = 0; kc < L_; kc += 32) {
      v16bf b = load_bt_bf16(Bb + kc, L_);
#pragma unroll
      for (int r4 = 0; r4 < 4; ++r4) {
        v16bf a = load_a_bf16(Ab + r4 * 16 * L_ + kc, L_);
        acc[r4] = wmma_bf16(a, b, acc[r4]);
      }
    }
  } else {
#pragma unroll
    for (int kc = 0; kc < L_; kc += 32) {
      v16bf b = load_bt_bf16_guard(Bb + kc, L_, 32, nvalid);
#pragma unroll
      for (int r4 = 0; r4 < 4; ++r4) {
        v16bf a = load_a_bf16(Ab + r4 * 16 * L_ + kc, L_);
        acc[r4] = wmma_bf16(a, b, acc[r4]);
      }
    }
  }
  const int col = n0 + nl;
  const bool colok = col < D_;
  const float bv = colok ? bias[col] : 0.0f;
#pragma unroll
  for (int r4 = 0; r4 < 4; ++r4) {
#pragma unroll
    for (int e = 0; e < 8; ++e) {
      if (colok) {
        const long row = rb * 64 + r4 * 16 + mhalf + e;
        zpred[row * D_ + col] = acc[r4][e] + bv;
      }
    }
  }
}

// ===========================================================================
extern "C" void kernel_launch(void* const* d_in, const int* in_sizes, int n_in,
                              void* d_out, int out_size, void* d_ws, size_t ws_size,
                              hipStream_t stream) {
  const float* z_seq  = (const float*)d_in[0];
  const float* eps    = (const float*)d_in[1];
  const float* proj_W = (const float*)d_in[2];
  const float* proj_b = (const float*)d_in[3];
  const float* Wih    = (const float*)d_in[4];
  const float* Whh    = (const float*)d_in[5];
  const float* bih    = (const float*)d_in[6];
  const float* bhh    = (const float*)d_in[7];
  const float* Wz     = (const float*)d_in[8];
  const float* bz     = (const float*)d_in[9];
  const float* oW1    = (const float*)d_in[10];
  const float* ob1    = (const float*)d_in[11];
  const float* oW2    = (const float*)d_in[12];
  const float* ob2    = (const float*)d_in[13];
  const float* gW1    = (const float*)d_in[14];
  const float* gb1    = (const float*)d_in[15];
  const float* gW2    = (const float*)d_in[16];
  const float* gb2    = (const float*)d_in[17];
  const float* rW     = (const float*)d_in[18];
  const float* rb     = (const float*)d_in[19];
  // (obs_len d_in[20] is a device scalar; OBS_=3 per reference)

  // ---- f32 workspace ----
  float* ws     = (float*)d_ws;
  float* xobs   = ws;                        // 49152
  float* part   = xobs  + 49152;             // 4*384*128 = 196608
  float* hA     = part  + 196608;            // 32768
  float* hB     = hA    + 32768;             // 32768
  float* zcurf  = hB    + 32768;             // 16384
  float* k1     = zcurf + 16384;             // 16384
  float* k2     = k1    + 16384;
  float* k3     = k2    + 16384;
  float* zmean  = k3    + 16384;             // 2816*128 = 360448
  // ---- bf16 workspace (16B-aligned: float count above is multiple of 4) ----
  __bf16* wb     = (__bf16*)(zmean + 360448);
  __bf16* projWb = wb;                       // 1035264
  __bf16* recWb  = projWb + 1035264;         // 1035264
  __bf16* Wihb   = recWb  + 1035264;         // 98304
  __bf16* Whhb   = Wihb   + 98304;           // 196608
  __bf16* Wzb    = Whhb   + 196608;          // 32768
  __bf16* W1b    = Wzb    + 32768;           // 32768
  __bf16* W2b    = W1b    + 32768;           // 32768
  __bf16* gW1b   = W2b    + 32768;           // 8192
  __bf16* gW2b   = gW1b   + 8192;            // 8192
  __bf16* zcurb  = gW2b   + 8192;            // 16384
  __bf16* zinb   = zcurb  + 16384;           // 16384
  __bf16* zsampb = zinb   + 16384;           // 360448

  float* zpred = (float*)d_out;              // 2816*8088
  float* lvout = zpred + (long)B_ * PRED_ * D_;   // 2816*128

  k_cvt<<<1024, 256, 0, stream>>>(proj_W, rW, Wih, Whh, Wz, oW1, oW2, gW1, gW2, wb);
  k_proj<<<dim3(24, 4), 256, 0, stream>>>(z_seq, projWb, part);
  k_proj_reduce<<<192, 256, 0, stream>>>(part, proj_b, xobs);
  k_recurrent<<<1, 1024, 0, stream>>>(xobs, Wihb, Whhb, bih, bhh, Wzb, bz,
                                      W1b, ob1, W2b, ob2,
                                      hA, hB, zcurf, zcurb, zinb, k1, k2, k3, zmean);
  k_gp_sample<<<22, 256, 0, stream>>>(zmean, eps, gW1b, gb1, gW2b, gb2,
                                      zsampb, lvout);
  k_rec<<<dim3(64, 44), 256, 0, stream>>>(zsampb, recWb, rb, zpred);
}
